// Attention_11897059410315
// MI455X (gfx1250) — compile-verified, matrix operand order checked by reading, not run on hardware
//
#include <hip/hip_runtime.h>
#include <hip/hip_bf16.h>

typedef _Float16 f16;
typedef __attribute__((ext_vector_type(16))) _Float16 v16h;
typedef __attribute__((ext_vector_type(8)))  _Float16 v8h;
typedef __attribute__((ext_vector_type(4)))  _Float16 v4h;
typedef __attribute__((ext_vector_type(8)))  float    v8f;

#define S_LEN   2048
#define D_MODEL 4096
#define NHEAD   32
#define NKV     8
#define HDIM    128
#define T_TOT   4096   // 2048 cache + 2048 new
#define KVCOLS  1024   // NKV * HDIM

__device__ __forceinline__ v16h comb16(v8h lo, v8h hi) {
    return __builtin_shufflevector(lo, hi, 0,1,2,3,4,5,6,7,8,9,10,11,12,13,14,15);
}

// ---------------------------------------------------------------------------
// Tiled GEMM: C[M,N] = A[M,K] * B[K,N], f32 weights converted to f16 on the
// fly.  Register-staged double-buffered LDS (ping-pong, 1 barrier/iter).
// Tile: BM=128, BN=128, BK=32; 256 threads = 8 waves (2x4); wave tile 64x32.
// ---------------------------------------------------------------------------
template <bool A_F16, bool ROPE, bool OUT_F32>
__global__ __launch_bounds__(256)
void gemm_wmma(const void* __restrict__ Aptr, const float* __restrict__ Bw,
               const float* __restrict__ rope, void* __restrict__ Cptr,
               int M, int N, int K, int ldC)
{
    constexpr int BM = 128, BN = 128, BK = 32;
    __shared__ __align__(16) f16 As[2][BM][BK + 8];   // 2 x 10 KB
    __shared__ __align__(16) f16 Bs[2][BN][BK + 8];   // transposed [n][k]

    const int tid  = threadIdx.x;
    const int wid  = tid >> 5;
    const int lane = tid & 31;
    const int wm   = wid >> 2;        // 0..1
    const int wn   = wid & 3;         // 0..3
    const int ml   = lane & 15;
    const int hi   = lane >> 4;

    const int m0 = blockIdx.y * BM;
    const int n0 = blockIdx.x * BN;

    // B staging: each thread owns a 4(k) x 4(n) sub-block
    const int kbB = (tid >> 5) * 4;   // 0..28
    const int nbB = (tid & 31) * 4;   // 0..124

    v8f acc[4][2] = {};

    float rav[4][4];                  // f32-A staging regs
    v8h   ra16[2];                    // f16-A staging regs
    float rbv[4][4];                  // B staging regs

    auto loadRegs = [&](int k0) {
        if constexpr (A_F16) {
            const f16* A = (const f16*)Aptr;
            #pragma unroll
            for (int i = 0; i < 2; ++i) {
                int f = tid + i * 256;  int r = f >> 2;  int c = (f & 3) * 8;
                ra16[i] = *(const v8h*)(A + (size_t)(m0 + r) * K + k0 + c);
            }
        } else {
            const float* A = (const float*)Aptr;
            #pragma unroll
            for (int i = 0; i < 4; ++i) {
                int f = tid + i * 256;  int r = f >> 3;  int c = (f & 7) * 4;
                float4 t = *(const float4*)(A + (size_t)(m0 + r) * K + k0 + c);
                rav[i][0] = t.x; rav[i][1] = t.y; rav[i][2] = t.z; rav[i][3] = t.w;
            }
        }
        #pragma unroll
        for (int j = 0; j < 4; ++j) {
            float4 t = *(const float4*)(Bw + (size_t)(k0 + kbB + j) * N + n0 + nbB);
            rbv[j][0] = t.x; rbv[j][1] = t.y; rbv[j][2] = t.z; rbv[j][3] = t.w;
        }
    };

    auto storeLDS = [&](int buf) {
        if constexpr (A_F16) {
            #pragma unroll
            for (int i = 0; i < 2; ++i) {
                int f = tid + i * 256;  int r = f >> 2;  int c = (f & 3) * 8;
                *(v8h*)&As[buf][r][c] = ra16[i];
            }
        } else {
            #pragma unroll
            for (int i = 0; i < 4; ++i) {
                int f = tid + i * 256;  int r = f >> 3;  int c = (f & 7) * 4;
                *(v4h*)&As[buf][r][c] =
                    (v4h){(f16)rav[i][0], (f16)rav[i][1], (f16)rav[i][2], (f16)rav[i][3]};
            }
        }
        #pragma unroll
        for (int m = 0; m < 4; ++m) {   // transposed: Bs[n][k], 8B DS stores
            *(v4h*)&Bs[buf][nbB + m][kbB] =
                (v4h){(f16)rbv[0][m], (f16)rbv[1][m], (f16)rbv[2][m], (f16)rbv[3][m]};
        }
    };

    loadRegs(0);
    int buf = 0;
    for (int k0 = 0; k0 < K; k0 += BK) {
        storeLDS(buf);
        __syncthreads();

        if (k0 + BK < K) loadRegs(k0 + BK);
        if (k0 + 2 * BK < K) {
            __builtin_prefetch(Bw + (size_t)(k0 + 2 * BK + kbB) * N + n0 + nbB, 0, 1);
            if constexpr (A_F16)
                __builtin_prefetch((const f16*)Aptr + (size_t)(m0 + (tid >> 2)) * K + k0 + 2 * BK, 0, 1);
            else
                __builtin_prefetch((const float*)Aptr + (size_t)(m0 + (tid >> 3)) * K + k0 + 2 * BK, 0, 1);
        }

        v16h af[4];
        #pragma unroll
        for (int mt = 0; mt < 4; ++mt) {
            int row = wm * 64 + mt * 16 + ml;
            v8h lo = *(const v8h*)&As[buf][row][hi * 8];
            v8h hh = *(const v8h*)&As[buf][row][16 + hi * 8];
            af[mt] = comb16(lo, hh);
        }
        #pragma unroll
        for (int nt = 0; nt < 2; ++nt) {
            int col = wn * 32 + nt * 16 + ml;
            v8h lo = *(const v8h*)&Bs[buf][col][hi * 8];
            v8h hh = *(const v8h*)&Bs[buf][col][16 + hi * 8];
            v16h bf = comb16(lo, hh);
            #pragma unroll
            for (int mt = 0; mt < 4; ++mt) {
                acc[mt][nt] = __builtin_amdgcn_wmma_f32_16x16x32_f16(
                    false, af[mt], false, bf, (short)0, acc[mt][nt], false, false);
            }
        }
        buf ^= 1;
    }

    // ---- epilogue ----
    #pragma unroll
    for (int mt = 0; mt < 4; ++mt) {
        #pragma unroll
        for (int nt = 0; nt < 2; ++nt) {
            #pragma unroll
            for (int v = 0; v < 8; ++v) {
                int r = m0 + wm * 64 + mt * 16 + v + 8 * hi;
                int c = n0 + wn * 32 + nt * 16 + ml;
                float val = acc[mt][nt][v];
                if (ROPE) val *= rope[(size_t)r * HDIM + (c & (HDIM - 1))];
                if (OUT_F32) ((float*)Cptr)[(size_t)r * ldC + c] = val;
                else         ((f16*) Cptr)[(size_t)r * ldC + c] = (f16)val;
            }
        }
    }
}

// ---------------------------------------------------------------------------
// Convert f32 KV cache -> f16 rows [0,2048) of contiguous K/V buffers.
// ---------------------------------------------------------------------------
__global__ __launch_bounds__(256)
void cvt_cache(const float* __restrict__ ck, const float* __restrict__ cv,
               f16* __restrict__ kf, f16* __restrict__ vf)
{
    size_t i = ((size_t)blockIdx.x * 256 + threadIdx.x) * 4;   // 2M elems total
    float4 a = *(const float4*)(ck + i);
    float4 b = *(const float4*)(cv + i);
    *(v4h*)(kf + i) = (v4h){(f16)a.x, (f16)a.y, (f16)a.z, (f16)a.w};
    *(v4h*)(vf + i) = (v4h){(f16)b.x, (f16)b.y, (f16)b.z, (f16)b.w};
}

// ---------------------------------------------------------------------------
// Flash attention: grid (16 m-tiles, 32 heads), 256 threads = 8 waves.
// K chunk staged via GLOBAL_LOAD_ASYNC_TO_LDS_B128 (ASYNCcnt path);
// V chunk register-transposed; Q fragments register-resident.
// ---------------------------------------------------------------------------
__global__ __launch_bounds__(256)
void flash_attn(const f16* __restrict__ Q, const f16* __restrict__ Kf,
                const f16* __restrict__ Vf, f16* __restrict__ Oout)
{
    __shared__ __align__(16) f16 Kl[64][HDIM + 8];   // [t][d], stride 136
    __shared__ __align__(16) f16 Vt[HDIM][64 + 8];   // [d][t], stride 72
    __shared__ __align__(16) f16 Pl[8][16][64 + 8];  // wave-private P strips

    const int tid  = threadIdx.x;
    const int w    = tid >> 5;
    const int lane = tid & 31;
    const int ml   = lane & 15;
    const int hi   = lane >> 4;

    const int h    = blockIdx.y;
    const int kvh  = h >> 2;              // N_REP = 4
    const int m0   = blockIdx.x * 128;
    const int srow = m0 + w * 16;         // wave's first query row

    // Q fragments in registers (rope already applied by projection GEMM)
    const f16* qrow = Q + (size_t)(srow + ml) * D_MODEL + h * HDIM;
    v16h aq[4];
    #pragma unroll
    for (int ks = 0; ks < 4; ++ks) {
        v8h lo = *(const v8h*)(qrow + ks * 32 + hi * 8);
        v8h hh = *(const v8h*)(qrow + ks * 32 + 16 + hi * 8);
        aq[ks] = comb16(lo, hh);
    }

    v8f Od[8] = {};
    float mrow[8], lrow[8];
    #pragma unroll
    for (int v = 0; v < 8; ++v) { mrow[v] = -1e30f; lrow[v] = 0.0f; }

    const float scale = 0.08838834764831845f;   // 1/sqrt(128)
    const int tEnd = min(T_TOT, 2048 + m0 + 128);   // causal chunk skip

    // V staging: each thread owns a 4(t) x 8(d) sub-block
    const int r4 = (tid >> 4) * 4;      // 0..60
    const int c8 = (tid & 15) * 8;      // 0..120

    for (int t0 = 0; t0 < tEnd; t0 += 64) {
        __syncthreads();

        // ---- K chunk: async DMA global -> LDS (b128 per lane) ----
        #pragma unroll
        for (int i = 0; i < 4; ++i) {
            int f = tid + i * 256;          // 0..1023
            int r = f >> 4;                 // 0..63 (t local)
            int c = (f & 15) * 8;           // 0..120 (d)
            unsigned dst = (unsigned)(size_t)&Kl[r][c];
            const f16* src = Kf + (size_t)(t0 + r) * KVCOLS + kvh * HDIM + c;
            asm volatile("global_load_async_to_lds_b128 %0, %1, off"
                         :: "v"(dst), "v"(src) : "memory");
        }

        // ---- V chunk: load rows, transpose in registers, v4h DS stores ----
        v8h va[4];
        #pragma unroll
        for (int j = 0; j < 4; ++j)
            va[j] = *(const v8h*)(Vf + (size_t)(t0 + r4 + j) * KVCOLS + kvh * HDIM + c8);
        #pragma unroll
        for (int jj = 0; jj < 8; ++jj) {
            *(v4h*)&Vt[c8 + jj][r4] = (v4h){va[0][jj], va[1][jj], va[2][jj], va[3][jj]};
        }
        if (t0 + 64 < tEnd)
            __builtin_prefetch(Vf + (size_t)(t0 + 64 + r4) * KVCOLS + kvh * HDIM + c8, 0, 1);

        asm volatile("s_wait_asynccnt 0x0" ::: "memory");
        __syncthreads();

        // ---- scores S = Q * K^T  (16 x 64 per wave) ----
        v8f sc[4];
        #pragma unroll
        for (int nt = 0; nt < 4; ++nt) {
            v8f s = {};
            int tl = nt * 16 + ml;
            #pragma unroll
            for (int ks = 0; ks < 4; ++ks) {
                v8h lo = *(const v8h*)&Kl[tl][ks * 32 + hi * 8];
                v8h hh = *(const v8h*)&Kl[tl][ks * 32 + 16 + hi * 8];
                v16h bf = comb16(lo, hh);
                s = __builtin_amdgcn_wmma_f32_16x16x32_f16(
                        false, aq[ks], false, bf, (short)0, s, false, false);
            }
            sc[nt] = s;
        }

        // ---- scale + analytic causal mask ----
        const bool needMask = (t0 + 63) > (2048 + srow);
        #pragma unroll
        for (int nt = 0; nt < 4; ++nt) {
            #pragma unroll
            for (int v = 0; v < 8; ++v) {
                float x = sc[nt][v] * scale;
                if (needMask) {
                    int t  = t0 + nt * 16 + ml;
                    int s_ = srow + v + 8 * hi;
                    if (t > 2048 + s_) x = -1e30f;
                }
                sc[nt][v] = x;
            }
        }

        // ---- online softmax row stats (rows live across 16 lanes) ----
        float alpha[8];
        #pragma unroll
        for (int v = 0; v < 8; ++v) {
            float mx = fmaxf(fmaxf(sc[0][v], sc[1][v]), fmaxf(sc[2][v], sc[3][v]));
            #pragma unroll
            for (int off = 8; off >= 1; off >>= 1)
                mx = fmaxf(mx, __shfl_xor(mx, off, 16));
            float mnew = fmaxf(mrow[v], mx);
            alpha[v] = __expf(mrow[v] - mnew);
            mrow[v] = mnew;
            float rs = 0.0f;
            #pragma unroll
            for (int nt = 0; nt < 4; ++nt) {
                float p = __expf(sc[nt][v] - mnew);
                sc[nt][v] = p;
                rs += p;
            }
            #pragma unroll
            for (int off = 8; off >= 1; off >>= 1)
                rs += __shfl_xor(rs, off, 16);
            lrow[v] = lrow[v] * alpha[v] + rs;
        }

        // ---- P -> LDS (f16) for A-fragment re-layout; rescale O ----
        #pragma unroll
        for (int nt = 0; nt < 4; ++nt)
            #pragma unroll
            for (int v = 0; v < 8; ++v)
                Pl[w][v + 8 * hi][nt * 16 + ml] = (f16)sc[nt][v];
        #pragma unroll
        for (int dt = 0; dt < 8; ++dt)
            #pragma unroll
            for (int v = 0; v < 8; ++v)
                Od[dt][v] *= alpha[v];

        // ---- O += P * V  (K = 64 in 2 steps) ----
        #pragma unroll
        for (int kt = 0; kt < 2; ++kt) {
            v8h plo = *(const v8h*)&Pl[w][ml][kt * 32 + hi * 8];
            v8h phh = *(const v8h*)&Pl[w][ml][kt * 32 + 16 + hi * 8];
            v16h ap = comb16(plo, phh);
            #pragma unroll
            for (int dt = 0; dt < 8; ++dt) {
                v8h lo = *(const v8h*)&Vt[dt * 16 + ml][kt * 32 + hi * 8];
                v8h hh = *(const v8h*)&Vt[dt * 16 + ml][kt * 32 + 16 + hi * 8];
                v16h bf = comb16(lo, hh);
                Od[dt] = __builtin_amdgcn_wmma_f32_16x16x32_f16(
                            false, ap, false, bf, (short)0, Od[dt], false, false);
            }
        }
    }

    // ---- finalize: divide by l, store f16 attn output [s][h*128+d] ----
    #pragma unroll
    for (int dt = 0; dt < 8; ++dt) {
        #pragma unroll
        for (int v = 0; v < 8; ++v) {
            int r = srow + v + 8 * hi;
            int c = h * HDIM + dt * 16 + ml;
            Oout[(size_t)r * D_MODEL + c] = (f16)(Od[dt][v] / lrow[v]);
        }
    }
}

// ---------------------------------------------------------------------------
extern "C" void kernel_launch(void* const* d_in, const int* in_sizes, int n_in,
                              void* d_out, int out_size, void* d_ws, size_t ws_size,
                              hipStream_t stream)
{
    const float* x       = (const float*)d_in[0];   // [2048,4096]
    const float* rope    = (const float*)d_in[1];   // [2048,128]
    // d_in[2] = mask (computed analytically in-kernel)
    const float* cache_k = (const float*)d_in[3];   // [2048,8,128]
    const float* cache_v = (const float*)d_in[4];
    const float* wq      = (const float*)d_in[5];   // [4096,4096]
    const float* wk      = (const float*)d_in[6];   // [4096,1024]
    const float* wv      = (const float*)d_in[7];
    const float* wo      = (const float*)d_in[8];   // [4096,4096]
    float* out           = (float*)d_out;           // [2048,4096]

    // workspace layout (f16): Q | Kfull | Vfull | attn   (48 MB total)
    f16* qf = (f16*)d_ws;                               // 2048*4096
    f16* kf = qf + (size_t)S_LEN * D_MODEL;             // 4096*1024
    f16* vf = kf + (size_t)T_TOT * KVCOLS;              // 4096*1024
    f16* at = vf + (size_t)T_TOT * KVCOLS;              // 2048*4096

    // 1) Q = rope * (x @ wq)   -> f16
    gemm_wmma<false, true, false><<<dim3(32, 16), 256, 0, stream>>>(
        x, wq, rope, qf, S_LEN, D_MODEL, D_MODEL, D_MODEL);
    // 2) K_new = rope * (x @ wk) -> rows [2048,4096) of kf
    gemm_wmma<false, true, false><<<dim3(8, 16), 256, 0, stream>>>(
        x, wk, rope, kf + (size_t)2048 * KVCOLS, S_LEN, KVCOLS, D_MODEL, KVCOLS);
    // 3) V_new = x @ wv -> rows [2048,4096) of vf
    gemm_wmma<false, false, false><<<dim3(8, 16), 256, 0, stream>>>(
        x, wv, nullptr, vf + (size_t)2048 * KVCOLS, S_LEN, KVCOLS, D_MODEL, KVCOLS);
    // 4) cache f32 -> f16 rows [0,2048)
    cvt_cache<<<2048, 256, 0, stream>>>(cache_k, cache_v, kf, vf);
    // 5) flash attention -> at (f16, [2048][4096])
    flash_attn<<<dim3(16, 32), 256, 0, stream>>>(qf, kf, vf, at);
    // 6) out = at @ wo -> f32
    gemm_wmma<true, false, true><<<dim3(32, 16), 256, 0, stream>>>(
        at, wo, nullptr, out, S_LEN, D_MODEL, D_MODEL, D_MODEL);
}
